// TripletLoss_50122268344327
// MI455X (gfx1250) — matrix operand order repaired and checked by reading, not compiled, not run on hardware
//
#include <hip/hip_runtime.h>
#include <hip/hip_bf16.h>
#include <math.h>

typedef __attribute__((ext_vector_type(16))) _Float16 v16h;
typedef __attribute__((ext_vector_type(8)))  _Float16 v8h;
typedef __attribute__((ext_vector_type(8)))  float    v8f;

#define TL_MARGIN 1.0f
#define N_MAX 512

// ---------------------------------------------------------------------------
// Kernel 1: convert f32 embeddings -> f16 (for WMMA A/B operands) and compute
// exact f32 squared row norms. One block per row. Deterministic tree reduce.
// ---------------------------------------------------------------------------
__global__ void __launch_bounds__(128)
tl_prep_kernel(const float* __restrict__ emb, _Float16* __restrict__ Xh,
               float* __restrict__ n2, int D) {
    __shared__ float red[128];
    const int row = blockIdx.x;
    const int t   = threadIdx.x;
    float acc = 0.0f;
    for (int k = t; k < D; k += blockDim.x) {
        const float x = emb[(size_t)row * D + k];
        Xh[(size_t)row * D + k] = (_Float16)x;
        acc += x * x;
    }
    red[t] = acc;
    __syncthreads();
    for (int s = blockDim.x >> 1; s > 0; s >>= 1) {
        if (t < s) red[t] += red[t + s];
        __syncthreads();
    }
    if (t == 0) n2[row] = red[0];
}

// ---------------------------------------------------------------------------
// Kernel 2: Gram matrix via v_wmma_f32_16x16x32_f16 + distance epilogue.
// One wave32 per 16x16 output tile; K accumulated in f32 across D/32 WMMAs.
//
// Operand packing follows the gfx1250 ISA layouts (CDNA5 ISA):
//   A (16x32 f16): lane L -> row L%16; halves [0..7] = K 8*(L/16)+[0..7],
//                  halves [8..15] = K 16+8*(L/16)+[0..7]   (2x 16B loads)
//   B (32x16 f16): lane L -> col L%16; same K layout as A (2x 16B loads).
//     B = X^T tile, so B column n is X row (col_base+n): contiguous in memory.
//   C/D (16x16 f32): c[v] = tile(M = v + 8*(L/16), N = L%16).
// ---------------------------------------------------------------------------
__global__ void __launch_bounds__(256)
tl_gram_dist_kernel(const _Float16* __restrict__ Xh, const float* __restrict__ n2,
                    float* __restrict__ dist, int N, int D) {
    const int lane        = threadIdx.x & 31;
    const int wave        = threadIdx.x >> 5;
    const int tilesPerRow = N >> 4;
    const int tile        = blockIdx.x * (blockDim.x >> 5) + wave;
    if (tile >= tilesPerRow * tilesPerRow) return;   // wave-uniform: EXEC all-1s
    const int ti = tile / tilesPerRow;
    const int tj = tile % tilesPerRow;
    const int lr = lane & 15;
    const int hi = lane >> 4;

    const _Float16* __restrict__ Arow = Xh + (size_t)(ti * 16 + lr) * D;
    const _Float16* __restrict__ Brow = Xh + (size_t)(tj * 16 + lr) * D;

    v8f c = {};
    #pragma unroll
    for (int kb = 0; kb < 128; kb += 32) {           // D == 128 for this problem
        union { v16h v; v8h h[2]; } a, b;
        a.h[0] = *(const v8h*)(Arow + kb + 8 * hi);
        a.h[1] = *(const v8h*)(Arow + kb + 16 + 8 * hi);
        b.h[0] = *(const v8h*)(Brow + kb + 8 * hi);
        b.h[1] = *(const v8h*)(Brow + kb + 16 + 8 * hi);
        c = __builtin_amdgcn_wmma_f32_16x16x32_f16(
                /*neg_a=*/false, a.v, /*neg_b=*/false, b.v,
                /*c_mod=*/(short)0, c, /*reuse_a=*/false, /*reuse_b=*/false);
    }

    const float n2j = n2[tj * 16 + lr];
    const int   col = tj * 16 + lr;
    #pragma unroll
    for (int v = 0; v < 8; ++v) {
        const int row = ti * 16 + hi * 8 + v;
        float sq = n2[row] + n2j - 2.0f * c[v];
        sq = fmaxf(sq, 0.0f);
        dist[(size_t)row * N + col] = (sq > 0.0f) ? sqrtf(sq) : 0.0f;
    }
}

// ---------------------------------------------------------------------------
// Kernel 3: per-anchor batch-all triplet reduction. One block per anchor.
// Fixed iteration order + shared-memory tree => deterministic partial sums.
// ---------------------------------------------------------------------------
__global__ void __launch_bounds__(256)
tl_triplet_kernel(const float* __restrict__ dist, const int* __restrict__ labels,
                  float* __restrict__ anchorTotal, unsigned int* __restrict__ anchorCnt,
                  int N) {
    __shared__ float         rowD[N_MAX];
    __shared__ unsigned char isNeg[N_MAX];
    __shared__ float         red[256];
    __shared__ int           cntS[2];
    const int i  = blockIdx.x;
    const int t  = threadIdx.x;
    const int li = labels[i];

    if (t < 2) cntS[t] = 0;
    __syncthreads();

    int myPos = 0, myNeg = 0;
    for (int j = t; j < N; j += blockDim.x) {
        rowD[j] = dist[(size_t)i * N + j];
        const int neg = (labels[j] != li);
        isNeg[j] = (unsigned char)neg;
        if (neg) myNeg++;
        else if (j != i) myPos++;
    }
    atomicAdd(&cntS[0], myPos);   // exact integer adds: deterministic
    atomicAdd(&cntS[1], myNeg);
    __syncthreads();
    const int np = cntS[0];
    const int nn = cntS[1];

    float s = 0.0f;
    for (int j = 0; j < N; ++j) {                    // positives (sparse)
        if (!isNeg[j] && j != i) {
            const float dpm = rowD[j] + TL_MARGIN;
            for (int k = t; k < N; k += blockDim.x) { // negatives
                if (isNeg[k]) {
                    const float v = dpm - rowD[k];
                    s += (v > 0.0f) ? v : 0.0f;
                }
            }
        }
    }
    red[t] = s;
    __syncthreads();
    for (int st = blockDim.x >> 1; st > 0; st >>= 1) {
        if (t < st) red[t] += red[t + st];
        __syncthreads();
    }
    if (t == 0) {
        anchorTotal[i] = red[0];
        anchorCnt[i]   = (unsigned int)np * (unsigned int)nn;
    }
}

// ---------------------------------------------------------------------------
// Kernel 4: final deterministic reduction + mean.
// ---------------------------------------------------------------------------
__global__ void __launch_bounds__(256)
tl_finalize_kernel(const float* __restrict__ anchorTotal,
                   const unsigned int* __restrict__ anchorCnt,
                   float* __restrict__ out, int N) {
    __shared__ float              redF[256];
    __shared__ unsigned long long redC[256];
    const int t = threadIdx.x;
    float s = 0.0f;
    unsigned long long c = 0ull;
    for (int i = t; i < N; i += blockDim.x) {
        s += anchorTotal[i];
        c += anchorCnt[i];
    }
    redF[t] = s; redC[t] = c;
    __syncthreads();
    for (int st = blockDim.x >> 1; st > 0; st >>= 1) {
        if (t < st) { redF[t] += redF[t + st]; redC[t] += redC[t + st]; }
        __syncthreads();
    }
    if (t == 0) out[0] = (redC[0] > 0ull) ? (redF[0] / (float)redC[0]) : 0.0f;
}

// ---------------------------------------------------------------------------
extern "C" void kernel_launch(void* const* d_in, const int* in_sizes, int n_in,
                              void* d_out, int out_size, void* d_ws, size_t ws_size,
                              hipStream_t stream) {
    const float* emb    = (const float*)d_in[0];
    const int*   labels = (const int*)d_in[1];
    const int N = in_sizes[1];              // 512
    const int D = in_sizes[0] / N;          // 128

    // Workspace layout (all 256B-aligned):
    char* ws = (char*)d_ws;
    size_t off = 0;
    _Float16* Xh = (_Float16*)(ws + off);
    off += (size_t)N * D * sizeof(_Float16);      off = (off + 255) & ~(size_t)255;
    float* n2 = (float*)(ws + off);
    off += (size_t)N * sizeof(float);             off = (off + 255) & ~(size_t)255;
    float* dist = (float*)(ws + off);
    off += (size_t)N * N * sizeof(float);         off = (off + 255) & ~(size_t)255;
    float* aTot = (float*)(ws + off);
    off += (size_t)N * sizeof(float);             off = (off + 255) & ~(size_t)255;
    unsigned int* aCnt = (unsigned int*)(ws + off);

    tl_prep_kernel<<<N, 128, 0, stream>>>(emb, Xh, n2, D);

    const int tilesPerRow = N / 16;
    const int tiles       = tilesPerRow * tilesPerRow;   // 1024
    const int wavesPerBlk = 8;                           // 256 threads
    tl_gram_dist_kernel<<<(tiles + wavesPerBlk - 1) / wavesPerBlk, 256, 0, stream>>>(
        Xh, n2, dist, N, D);

    tl_triplet_kernel<<<N, 256, 0, stream>>>(dist, labels, aTot, aCnt, N);

    tl_finalize_kernel<<<1, 256, 0, stream>>>(aTot, aCnt, (float*)d_out, N);
}
